// IntraPatchGraphAggregator_43301860278559
// MI455X (gfx1250) — matrix operand order repaired and checked by reading, not compiled, not run on hardware
//
#include <hip/hip_runtime.h>
#include <hip/hip_bf16.h>
#include <math.h>

#define DMODEL 768
#define LSEQ   512
#define NBATCH 64
#define NHEADS 12
#define DHEAD  64
#define NEGV   -1e9f
#define WELEM  (DMODEL * DMODEL)   // 589824 elements per weight matrix

typedef __attribute__((ext_vector_type(16))) __bf16 v16bf;
typedef __attribute__((ext_vector_type(8)))  __bf16 v8bf;
typedef __attribute__((ext_vector_type(4)))  __bf16 v4bf;
typedef __attribute__((ext_vector_type(8)))  float  v8f;

static __device__ inline v16bf cat8(v8bf lo, v8bf hi) {
  v16bf r;
#pragma unroll
  for (int i = 0; i < 8; ++i) { r[i] = lo[i]; r[i + 8] = hi[i]; }
  return r;
}

static __device__ inline v8bf cvt8(float4 a, float4 b) {
  v8bf r;
  r[0] = (__bf16)a.x; r[1] = (__bf16)a.y; r[2] = (__bf16)a.z; r[3] = (__bf16)a.w;
  r[4] = (__bf16)b.x; r[5] = (__bf16)b.y; r[6] = (__bf16)b.z; r[7] = (__bf16)b.w;
  return r;
}

static __device__ inline v8bf ld8(const __bf16* p) { return *(const v8bf*)p; }

static __device__ inline v8f wmma_bf16(v16bf a, v16bf b, v8f c) {
  return __builtin_amdgcn_wmma_f32_16x16x32_bf16(false, a, false, b, (short)0, c,
                                                 false, false);
}

// LDS byte offset of a __shared__-derived generic pointer (aperture scheme:
// low 32 bits of the generic address are the wave-relative LDS offset).
static __device__ inline unsigned lds_off(const void* p) {
  return (unsigned)(uintptr_t)p;
}

// Async copy 16 bytes/lane: global (bf16, pre-converted) -> LDS.  ASYNCcnt.
static __device__ inline void async_cp16(unsigned lds, const __bf16* g) {
  asm volatile("global_load_async_to_lds_b128 %0, %1, off"
               :: "v"(lds), "v"(g) : "memory");
}
static __device__ inline void wait_async_le1() {
  asm volatile("s_wait_asynccnt 0x1" ::: "memory");
}
static __device__ inline void wait_async_le0() {
  asm volatile("s_wait_asynccnt 0x0" ::: "memory");
}

// ---------------------------------------------------------------------------
// Kernel 0: one-shot fp32 -> bf16 conversion of the four weight matrices so
// the GEMMs can stage them with async-to-LDS (raw byte copies).
// grid = (WELEM/1024, 4), block = 256, 4 elements/thread.
// ---------------------------------------------------------------------------
__global__ __launch_bounds__(256) void wcvt(
    const float* __restrict__ Wq, const float* __restrict__ Wk,
    const float* __restrict__ Wv, const float* __restrict__ Wo,
    __bf16* __restrict__ wall)
{
  const int m = blockIdx.y;
  const float* src = (m == 0) ? Wq : (m == 1) ? Wk : (m == 2) ? Wv : Wo;
  __bf16* dst = wall + (size_t)m * WELEM;
  const int i = (blockIdx.x * 256 + threadIdx.x) * 4;
  float4 v = *(const float4*)(src + i);
  v4bf o;
  o[0] = (__bf16)v.x; o[1] = (__bf16)v.y; o[2] = (__bf16)v.z; o[3] = (__bf16)v.w;
  *(v4bf*)(dst + i) = o;
}

// ---------------------------------------------------------------------------
// Kernel 1: fused QKV projection.  C = tokens @ W^T + b   (M=32768, N=K=768)
// grid = (M/128, 768/64, 3)  block = 256 (8 waves).  Wave computes 16x64.
// Weight tiles (bf16) staged by double-buffered GLOBAL_LOAD_ASYNC_TO_LDS.
// q,k stored row-major bf16 [M][768]; v stored transposed [n][h][dh][l] bf16.
// ---------------------------------------------------------------------------
__global__ __launch_bounds__(256) void qkv_gemm(
    const float* __restrict__ tokens, const __bf16* __restrict__ wall,
    const float* __restrict__ bq, const float* __restrict__ bk_,
    const float* __restrict__ bv_,
    __bf16* __restrict__ qb, __bf16* __restrict__ kb, __bf16* __restrict__ vT)
{
  __shared__ __bf16 WT[2][64][32];
  const int tid = threadIdx.x;
  const int lane = tid & 31, wid = tid >> 5;
  const int z = blockIdx.z;
  const __bf16* W   = wall + (size_t)z * WELEM;
  const float* bias = (z == 0) ? bq : (z == 1) ? bk_ : bv_;

  const int r0  = blockIdx.x * 128 + wid * 16;
  const int nb0 = blockIdx.y * 64;
  const int khalf = (lane >> 4) * 8;
  const int bk0   = (lane >> 4) * 16;
  const float* aptr = tokens + (size_t)(r0 + (lane & 15)) * DMODEL;
  const int lnfill = tid >> 2;        // 0..63 : local N row of W tile
  const int lkfill = (tid & 3) * 8;   // 0,8,16,24
  const __bf16* wsrc = W + (size_t)(nb0 + lnfill) * DMODEL + lkfill;
  const unsigned ldsA = lds_off(&WT[0][lnfill][lkfill]);
  const unsigned ldsB = lds_off(&WT[1][lnfill][lkfill]);

  v8f acc0 = {}, acc1 = {}, acc2 = {}, acc3 = {};

  async_cp16(ldsA, wsrc);                       // prefetch K-step 0
  for (int step = 0; step < DMODEL / 32; ++step) {
    const int kk = step * 32;
    if (step + 1 < DMODEL / 32) {
      async_cp16((step & 1) ? ldsA : ldsB, wsrc + kk + 32);
      wait_async_le1();                         // step's tile landed
    } else {
      wait_async_le0();
    }
    __syncthreads();

    __builtin_prefetch(aptr + kk + 64, 0, 3);   // pull next A rows toward WGP
    float4 a0 = *(const float4*)(aptr + kk + khalf);
    float4 a1 = *(const float4*)(aptr + kk + khalf + 4);
    float4 a2 = *(const float4*)(aptr + kk + khalf + 16);
    float4 a3 = *(const float4*)(aptr + kk + khalf + 20);
    v16bf af = cat8(cvt8(a0, a1), cvt8(a2, a3));

    const __bf16 (*Wc)[32] = WT[step & 1];
    v16bf b0 = cat8(ld8(&Wc[(lane & 15) +  0][khalf]), ld8(&Wc[(lane & 15) +  0][khalf + 16]));
    v16bf b1 = cat8(ld8(&Wc[(lane & 15) + 16][khalf]), ld8(&Wc[(lane & 15) + 16][khalf + 16]));
    v16bf b2 = cat8(ld8(&Wc[(lane & 15) + 32][khalf]), ld8(&Wc[(lane & 15) + 32][khalf + 16]));
    v16bf b3 = cat8(ld8(&Wc[(lane & 15) + 48][khalf]), ld8(&Wc[(lane & 15) + 48][khalf + 16]));

    acc0 = wmma_bf16(af, b0, acc0);
    acc1 = wmma_bf16(af, b1, acc1);
    acc2 = wmma_bf16(af, b2, acc2);
    acc3 = wmma_bf16(af, b3, acc3);
    __syncthreads();
  }

  const int mrow0 = r0 + 8 * (lane >> 4);
  v8f accs[4] = {acc0, acc1, acc2, acc3};
#pragma unroll
  for (int t = 0; t < 4; ++t) {
    const int ncol = nb0 + t * 16 + (lane & 15);
    const float bval = bias[ncol];
    v8f c = accs[t];
    if (z < 2) {
      __bf16* dst = (z == 0) ? qb : kb;
#pragma unroll
      for (int v = 0; v < 8; ++v)
        dst[(size_t)(mrow0 + v) * DMODEL + ncol] = (__bf16)(c[v] + bval);
    } else {
      const int nIdx = mrow0 >> 9, l0 = mrow0 & 511;
      const int hh = ncol >> 6, dh = ncol & 63;
      v8bf pk;
#pragma unroll
      for (int v = 0; v < 8; ++v) pk[v] = (__bf16)(c[v] + bval);
      *(v8bf*)(vT + (((size_t)nIdx * NHEADS + hh) * DHEAD + dh) * LSEQ + l0) = pk;
    }
  }
}

// ---------------------------------------------------------------------------
// Kernel 2: attention core.  One block per (64 q-rows, head, batch).
// S tile (64x512 fp32) lives in LDS.  Softmax in place, then O = P @ V.
// ---------------------------------------------------------------------------
__global__ __launch_bounds__(256) void attn_kernel(
    const __bf16* __restrict__ qb, const __bf16* __restrict__ kb,
    const __bf16* __restrict__ vT,
    const float* __restrict__ ts, const float* __restrict__ mask,
    __bf16* __restrict__ ao)
{
  __shared__ float S[64 * 512];
  __shared__ float lts[512];
  __shared__ float lbad[512];
  const int tid = threadIdx.x, lane = tid & 31, wid = tid >> 5;
  const int q0 = blockIdx.x * 64, h = blockIdx.y, n = blockIdx.z;

  for (int i = tid; i < 512; i += 256) {
    lts[i]  = ts[(size_t)n * LSEQ + i];
    lbad[i] = (mask[(size_t)n * LSEQ + i] <= 0.f) ? 1.f : 0.f;
  }
  __syncthreads();

  // ---- Phase 1: S = Q K^T * scale + time-decay bias, masked ----
  const int mt    = wid & 3;
  const int khalf = (lane >> 4) * 8;
  const int bk0   = (lane >> 4) * 16;
  const int qrow  = q0 + mt * 16 + (lane & 15);
  const __bf16* qp = qb + ((size_t)n * LSEQ + qrow) * DMODEL + h * DHEAD;
  v16bf a0 = cat8(ld8(qp + khalf),      ld8(qp + 16 + khalf));
  v16bf a1 = cat8(ld8(qp + 32 + khalf), ld8(qp + 48 + khalf));

  for (int nt = (wid >> 2); nt < 32; nt += 2) {
    const int kcol = nt * 16 + (lane & 15);
    const __bf16* kp = kb + ((size_t)n * LSEQ + kcol) * DMODEL + h * DHEAD;
    v16bf b0 = cat8(ld8(kp + khalf),      ld8(kp + khalf + 16));
    v16bf b1 = cat8(ld8(kp + 32 + khalf), ld8(kp + 32 + khalf + 16));
    v8f c = {};
    c = wmma_bf16(a0, b0, c);
    c = wmma_bf16(a1, b1, c);
    const float tk  = lts[kcol];
    const float bad = lbad[kcol];
    const int lrow0 = mt * 16 + 8 * (lane >> 4);
#pragma unroll
    for (int v = 0; v < 8; ++v) {
      float sv = c[v] * 0.125f;
      float d  = fabsf(lts[q0 + lrow0 + v] - tk);
      sv += __logf(__expf(-d * (1.0f / 300.0f)) + 1e-12f);
      if (bad > 0.f) sv = NEGV;
      S[(lrow0 + v) * 512 + kcol] = sv;
    }
  }
  __syncthreads();

  // ---- Phase 2: row softmax in place (8 rows per wave) ----
  for (int r = wid * 8; r < wid * 8 + 8; ++r) {
    float* row = &S[r * 512];
    float mx = -3.4e38f;
    for (int c2 = lane; c2 < 512; c2 += 32) mx = fmaxf(mx, row[c2]);
#pragma unroll
    for (int o = 16; o > 0; o >>= 1) mx = fmaxf(mx, __shfl_xor(mx, o, 32));
    float sum = 0.f;
    for (int c2 = lane; c2 < 512; c2 += 32) {
      float e = __expf(row[c2] - mx);
      row[c2] = e;
      sum += e;
    }
#pragma unroll
    for (int o = 16; o > 0; o >>= 1) sum += __shfl_xor(sum, o, 32);
    const float inv = 1.0f / sum;
    for (int c2 = lane; c2 < 512; c2 += 32) row[c2] *= inv;
  }
  __syncthreads();

  // ---- Phase 3: O = P @ V  (A from LDS fp32 -> bf16, B from transposed V) ----
#pragma unroll
  for (int rep = 0; rep < 2; ++rep) {
    const int id  = wid + rep * 8;       // 0..15
    const int m2  = id & 3, ntd = id >> 2;
    const __bf16* vp =
        vT + (((size_t)n * NHEADS + h) * DHEAD + ntd * 16 + (lane & 15)) * LSEQ;
    v8f acc = {};
    for (int k = 0; k < 512; k += 32) {
      const float* sp = &S[(m2 * 16 + (lane & 15)) * 512 + k + khalf];
      float4 f0 = *(const float4*)(sp);
      float4 f1 = *(const float4*)(sp + 4);
      float4 f2 = *(const float4*)(sp + 16);
      float4 f3 = *(const float4*)(sp + 20);
      v16bf af  = cat8(cvt8(f0, f1), cvt8(f2, f3));
      v16bf bf_ = cat8(ld8(vp + k + khalf), ld8(vp + k + khalf + 16));
      acc = wmma_bf16(af, bf_, acc);
    }
    const int dhc = ntd * 16 + (lane & 15);
    const int l0  = q0 + m2 * 16 + 8 * (lane >> 4);
#pragma unroll
    for (int v = 0; v < 8; ++v)
      ao[((size_t)n * LSEQ + l0 + v) * DMODEL + h * DHEAD + dhc] = (__bf16)acc[v];
  }
}

// ---------------------------------------------------------------------------
// Kernel 3: out = (ao @ Wo^T + bo) * mask   -> fp32 [32768][768]
// Same async-staged skeleton; A rows are already bf16.
// ---------------------------------------------------------------------------
__global__ __launch_bounds__(256) void wo_gemm(
    const __bf16* __restrict__ ao, const __bf16* __restrict__ wob,
    const float* __restrict__ bo, const float* __restrict__ mask,
    float* __restrict__ outp)
{
  __shared__ __bf16 WT[2][64][32];
  const int tid = threadIdx.x, lane = tid & 31, wid = tid >> 5;
  const int r0  = blockIdx.x * 128 + wid * 16;
  const int nb0 = blockIdx.y * 64;
  const int khalf = (lane >> 4) * 8;
  const int bk0   = (lane >> 4) * 16;
  const __bf16* aptr = ao + (size_t)(r0 + (lane & 15)) * DMODEL;
  const int lnfill = tid >> 2, lkfill = (tid & 3) * 8;
  const __bf16* wsrc = wob + (size_t)(nb0 + lnfill) * DMODEL + lkfill;
  const unsigned ldsA = lds_off(&WT[0][lnfill][lkfill]);
  const unsigned ldsB = lds_off(&WT[1][lnfill][lkfill]);

  v8f acc0 = {}, acc1 = {}, acc2 = {}, acc3 = {};

  async_cp16(ldsA, wsrc);
  for (int step = 0; step < DMODEL / 32; ++step) {
    const int kk = step * 32;
    if (step + 1 < DMODEL / 32) {
      async_cp16((step & 1) ? ldsA : ldsB, wsrc + kk + 32);
      wait_async_le1();
    } else {
      wait_async_le0();
    }
    __syncthreads();

    __builtin_prefetch(aptr + kk + 64, 0, 3);
    v16bf af = cat8(ld8(aptr + kk + khalf), ld8(aptr + kk + 16 + khalf));

    const __bf16 (*Wc)[32] = WT[step & 1];
    v16bf b0 = cat8(ld8(&Wc[(lane & 15) +  0][khalf]), ld8(&Wc[(lane & 15) +  0][khalf + 16]));
    v16bf b1 = cat8(ld8(&Wc[(lane & 15) + 16][khalf]), ld8(&Wc[(lane & 15) + 16][khalf + 16]));
    v16bf b2 = cat8(ld8(&Wc[(lane & 15) + 32][khalf]), ld8(&Wc[(lane & 15) + 32][khalf + 16]));
    v16bf b3 = cat8(ld8(&Wc[(lane & 15) + 48][khalf]), ld8(&Wc[(lane & 15) + 48][khalf + 16]));

    acc0 = wmma_bf16(af, b0, acc0);
    acc1 = wmma_bf16(af, b1, acc1);
    acc2 = wmma_bf16(af, b2, acc2);
    acc3 = wmma_bf16(af, b3, acc3);
    __syncthreads();
  }

  const int mrow0 = r0 + 8 * (lane >> 4);
  float mk[8];
#pragma unroll
  for (int v = 0; v < 8; ++v) mk[v] = mask[mrow0 + v];
  v8f accs[4] = {acc0, acc1, acc2, acc3};
#pragma unroll
  for (int t = 0; t < 4; ++t) {
    const int ncol = nb0 + t * 16 + (lane & 15);
    const float bval = bo[ncol];
    v8f c = accs[t];
#pragma unroll
    for (int v = 0; v < 8; ++v)
      outp[(size_t)(mrow0 + v) * DMODEL + ncol] = (c[v] + bval) * mk[v];
  }
}

// ---------------------------------------------------------------------------
// Kernel 4: pooled readout attention (tiny).  grid = (H, N), 64 threads.
// ---------------------------------------------------------------------------
__global__ __launch_bounds__(64) void pool_kernel(
    const float* __restrict__ outp, const float* __restrict__ readout,
    const float* __restrict__ mask, float* __restrict__ dout)
{
  __shared__ float sc[512];
  __shared__ float red[64];
  const int tid = threadIdx.x;
  const int h = blockIdx.x, n = blockIdx.y;
  const float* ro = readout + h * DHEAD;

  for (int l = tid; l < 512; l += 64) {
    const float* op = outp + ((size_t)n * LSEQ + l) * DMODEL + h * DHEAD;
    float dot = 0.f;
    for (int d = 0; d < DHEAD; d += 4) {
      float4 r4 = *(const float4*)(ro + d);
      float4 o4 = *(const float4*)(op + d);
      dot += r4.x * o4.x + r4.y * o4.y + r4.z * o4.z + r4.w * o4.w;
    }
    float val = dot * 0.125f;
    if (mask[(size_t)n * LSEQ + l] <= 0.f) val = NEGV;
    sc[l] = val;
  }
  __syncthreads();

  float mx = -3.4e38f;
  for (int l = tid; l < 512; l += 64) mx = fmaxf(mx, sc[l]);
  red[tid] = mx;
  __syncthreads();
  if (tid < 32) red[tid] = fmaxf(red[tid], red[tid + 32]);
  __syncthreads();
  if (tid == 0) {
    float m = red[0];
    for (int i = 1; i < 32; ++i) m = fmaxf(m, red[i]);
    red[0] = m;
  }
  __syncthreads();
  mx = red[0];
  __syncthreads();

  float sum = 0.f;
  for (int l = tid; l < 512; l += 64) {
    float e = __expf(sc[l] - mx);
    sc[l] = e;
    sum += e;
  }
  __syncthreads();
  red[tid] = sum;
  __syncthreads();
  if (tid < 32) red[tid] += red[tid + 32];
  __syncthreads();
  if (tid == 0) {
    float s = 0.f;
    for (int i = 0; i < 32; ++i) s += red[i];
    red[0] = s;
  }
  __syncthreads();
  const float inv = 1.0f / red[0];

  float accp = 0.f;
  const float* base = outp + (size_t)n * LSEQ * DMODEL + h * DHEAD + tid;
  for (int l = 0; l < 512; ++l) accp += sc[l] * base[(size_t)l * DMODEL];
  dout[(size_t)n * DMODEL + h * DHEAD + tid] = accp * inv;
}

// ---------------------------------------------------------------------------
extern "C" void kernel_launch(void* const* d_in, const int* in_sizes, int n_in,
                              void* d_out, int out_size, void* d_ws, size_t ws_size,
                              hipStream_t stream)
{
  const float* tokens  = (const float*)d_in[0];
  const float* ts      = (const float*)d_in[1];
  const float* mask    = (const float*)d_in[2];
  const float* Wq      = (const float*)d_in[3];
  const float* bq      = (const float*)d_in[4];
  const float* Wk      = (const float*)d_in[5];
  const float* bk      = (const float*)d_in[6];
  const float* Wv      = (const float*)d_in[7];
  const float* bv      = (const float*)d_in[8];
  const float* Wo      = (const float*)d_in[9];
  const float* bo      = (const float*)d_in[10];
  const float* readout = (const float*)d_in[11];
  float* dout = (float*)d_out;

  const size_t M    = (size_t)NBATCH * LSEQ;  // 32768
  const size_t nelm = M * DMODEL;
  __bf16* qb   = (__bf16*)d_ws;
  __bf16* kb   = qb + nelm;
  __bf16* vT   = kb + nelm;
  __bf16* ao   = vT + nelm;
  float*  outp = (float*)(ao + nelm);
  __bf16* wall = (__bf16*)(outp + nelm);      // 4 x 768 x 768 bf16

  wcvt<<<dim3(WELEM / 1024, 4), 256, 0, stream>>>(Wq, Wk, Wv, Wo, wall);
  qkv_gemm<<<dim3((unsigned)(M / 128), DMODEL / 64, 3), 256, 0, stream>>>(
      tokens, wall, bq, bk, bv, qb, kb, vT);
  attn_kernel<<<dim3(LSEQ / 64, NHEADS, NBATCH), 256, 0, stream>>>(
      qb, kb, vT, ts, mask, ao);
  wo_gemm<<<dim3((unsigned)(M / 128), DMODEL / 64), 256, 0, stream>>>(
      ao, wall + (size_t)3 * WELEM, bo, mask, outp);
  pool_kernel<<<dim3(NHEADS, NBATCH), 64, 0, stream>>>(
      outp, readout, mask, dout);
}
